// MultiHeadAttentionAPE_11854109737274
// MI455X (gfx1250) — matrix operand order repaired and checked by reading, not compiled, not run on hardware
//
#include <hip/hip_runtime.h>

// Problem constants (B=2, S=2048, D=1024, H=16, HD=64)
#define S_LEN 2048
#define D_DIM 1024
#define NH 16
#define HD_DIM 64
#define BH 32          // B*H
#define M_ROWS 4096    // B*S

typedef __bf16 bf16_t;
typedef __attribute__((ext_vector_type(8)))  __bf16 bf16x8;
typedef __attribute__((ext_vector_type(16))) __bf16 bf16x16;
typedef __attribute__((ext_vector_type(8)))  float  f32x8;

static __device__ __forceinline__ bf16_t f2bf(float f) {
  unsigned u = __builtin_bit_cast(unsigned, f);
  u += 0x7FFFu + ((u >> 16) & 1u);           // round-to-nearest-even
  unsigned short h = (unsigned short)(u >> 16);
  return __builtin_bit_cast(bf16_t, h);
}

// A-matrix 16x32 bf16 fragment (ISA 7.12.2): lane holds row m=lane&15,
// element j<8 -> k = kb+j, element j>=8 -> k = kb+16+(j-8), kb=(lane>>4)*8.
static __device__ __forceinline__ bf16x16 load_a_frag(const bf16_t* row, int lane) {
  const int kb = (lane >> 4) << 3;
  bf16x8 lo = *(const bf16x8*)(row + kb);
  bf16x8 hi = *(const bf16x8*)(row + kb + 16);
  bf16x16 r;
#pragma unroll
  for (int i = 0; i < 8; ++i) { r[i] = lo[i]; r[i + 8] = hi[i]; }
  return r;
}

// B-matrix 32x16 bf16 fragment: lane holds column n=lane&15; element j<8 ->
// k = kb+j, element j>=8 -> k = kb+16+(j-8), kb=(lane>>4)*8 (same as A).
static __device__ __forceinline__ bf16x16 load_b_frag(const bf16_t* row, int lane) {
  const int kb = (lane >> 4) << 3;
  bf16x8 lo = *(const bf16x8*)(row + kb);
  bf16x8 hi = *(const bf16x8*)(row + kb + 16);
  bf16x16 r;
#pragma unroll
  for (int i = 0; i < 8; ++i) { r[i] = lo[i]; r[i + 8] = hi[i]; }
  return r;
}

static __device__ __forceinline__ f32x8 wmma_bf16(bf16x16 a, bf16x16 b, f32x8 c) {
  return __builtin_amdgcn_wmma_f32_16x16x32_bf16(false, a, false, b, (short)0, c,
                                                 false, false);
}

// CDNA5 async copy global->LDS (ASYNCcnt-tracked); 16B per lane.
static __device__ __forceinline__ void async_load_b128(unsigned lds_off,
                                                       const bf16_t* gptr) {
  unsigned long long ga = (unsigned long long)(size_t)gptr;
  asm volatile("global_load_async_to_lds_b128 %0, %1, off"
               :: "v"(lds_off), "v"(ga) : "memory");
}
#define WAIT_ASYNCCNT_4() asm volatile("s_wait_asynccnt 0x4" ::: "memory")
#define WAIT_ASYNCCNT_0() asm volatile("s_wait_asynccnt 0x0" ::: "memory")

// Shared pipelined GEMM core: wave tile 32x64, K = D_DIM, unrolled x2 with
// ping-pong fragment sets so double-buffering costs zero register moves.
static __device__ __forceinline__ void gemm_core_32x64(
    const bf16_t* __restrict__ arow0, const bf16_t* __restrict__ arow1,
    const bf16_t* const brow[4], int lane, f32x8 acc[2][4]) {
  bf16x16 aA[2], bA[4], aB[2], bB[4];
  aA[0] = load_a_frag(arow0, lane);
  aA[1] = load_a_frag(arow1, lane);
#pragma unroll
  for (int t = 0; t < 4; ++t) bA[t] = load_b_frag(brow[t], lane);

  for (int k0 = 0; k0 < D_DIM; k0 += 64) {
    const int k1 = (k0 + 32) & (D_DIM - 1);
    aB[0] = load_a_frag(arow0 + k1, lane);
    aB[1] = load_a_frag(arow1 + k1, lane);
#pragma unroll
    for (int t = 0; t < 4; ++t) bB[t] = load_b_frag(brow[t] + k1, lane);
#pragma unroll
    for (int t = 0; t < 4; ++t) {
      acc[0][t] = wmma_bf16(aA[0], bA[t], acc[0][t]);
      acc[1][t] = wmma_bf16(aA[1], bA[t], acc[1][t]);
    }
    const int k2 = (k0 + 64) & (D_DIM - 1);  // wraps to 0 on final prefetch
    aA[0] = load_a_frag(arow0 + k2, lane);
    aA[1] = load_a_frag(arow1 + k2, lane);
#pragma unroll
    for (int t = 0; t < 4; ++t) bA[t] = load_b_frag(brow[t] + k2, lane);
#pragma unroll
    for (int t = 0; t < 4; ++t) {
      acc[0][t] = wmma_bf16(aB[0], bB[t], acc[0][t]);
      acc[1][t] = wmma_bf16(aB[1], bB[t], acc[1][t]);
    }
  }
}

// ---------------------------------------------------------------- pre-pass ---
__global__ __launch_bounds__(256) void cvt_bf16_kernel(const float* __restrict__ in,
                                                       bf16_t* __restrict__ out, int n) {
  int i = blockIdx.x * blockDim.x + threadIdx.x;
  if (i < n) out[i] = f2bf(in[i]);
}

// in: [K][N] f32 row-major -> out: [N][K] bf16. Grid (K/32, N/8), block 256.
// lane -> k (coalesced writes), wave -> n; no integer division.
__global__ __launch_bounds__(256) void transpose_bf16_kernel(const float* __restrict__ in,
                                                             bf16_t* __restrict__ out,
                                                             int K, int N) {
  const int k = blockIdx.x * 32 + (threadIdx.x & 31);
  const int n = blockIdx.y * 8 + (threadIdx.x >> 5);
  out[(size_t)n * K + k] = f2bf(in[(size_t)k * N + n]);
}

// ---------------------------------------------------------------- QKV GEMM ---
// e -> h = e/192, type = (e%192)/64, hd = e%64. V stored transposed [bh][hd][s].
__global__ __launch_bounds__(128) void qkv_gemm_kernel(
    const bf16_t* __restrict__ xbf,   // [4096][1024]
    const bf16_t* __restrict__ wqt,   // [3072][1024]  (= W_qkv^T, bf16)
    const float*  __restrict__ bias,  // [3072]
    bf16_t* __restrict__ qb, bf16_t* __restrict__ kb, bf16_t* __restrict__ vtb) {
  const int lane = threadIdx.x & 31;
  const int wave = threadIdx.x >> 5;
  const int row0 = blockIdx.x * 128 + wave * 32;
  const int col0 = blockIdx.y * 64;
  const int m = lane & 15;
  const int nlo = lane & 15;
  const int mtop = (lane >> 4) << 3;

  const bf16_t* arow0 = xbf + (size_t)(row0 + m) * D_DIM;
  const bf16_t* arow1 = xbf + (size_t)(row0 + 16 + m) * D_DIM;
  const bf16_t* brow[4];
#pragma unroll
  for (int t = 0; t < 4; ++t)
    brow[t] = wqt + (size_t)(col0 + t * 16 + nlo) * D_DIM;

  f32x8 acc[2][4] = {};
  gemm_core_32x64(arow0, arow1, brow, lane, acc);

#pragma unroll
  for (int t = 0; t < 4; ++t) {
    const int e = col0 + t * 16 + nlo;
    const float bv = bias[e];
    const int h = e / 192, rr = e - h * 192, type = rr >> 6, hd = rr & 63;
#pragma unroll
    for (int r = 0; r < 2; ++r) {
#pragma unroll
      for (int i = 0; i < 8; ++i) {
        const int mr = row0 + r * 16 + mtop + i;
        const int bb = mr >> 11, s = mr & (S_LEN - 1);
        const bf16_t v = f2bf(acc[r][t][i] + bv);
        const size_t bh = (size_t)(bb * NH + h);
        if (type == 0)      qb[(bh * S_LEN + s) * HD_DIM + hd] = v;
        else if (type == 1) kb[(bh * S_LEN + s) * HD_DIM + hd] = v;
        else                vtb[(bh * HD_DIM + hd) * S_LEN + s] = v;
      }
    }
  }
}

// --------------------------------------------------------------- attention ---
// 4 waves = 64 query rows of one (b,h); block-uniform k loop so K/V tiles are
// staged cooperatively in LDS via double-buffered async-to-LDS copies.
__global__ __launch_bounds__(128) void attn_kernel(
    const bf16_t* __restrict__ qb,   // [BH][S][64]
    const bf16_t* __restrict__ kbm,  // [BH][S][64]
    const bf16_t* __restrict__ vtb,  // [BH][64][S]
    bf16_t* __restrict__ ob) {       // [BH][S][64] flat == reference reshape
  __shared__ __attribute__((aligned(16))) bf16_t ktile[2][32][64];  // 8 KB
  __shared__ __attribute__((aligned(16))) bf16_t vtile[2][64][32];  // 8 KB
  __shared__ __attribute__((aligned(16))) bf16_t psh[4][16][32];    // 4 KB
  const int tid  = threadIdx.x;
  const int lane = tid & 31;
  const int wave = tid >> 5;
  const int bh   = blockIdx.x >> 5;
  const int qblk = blockIdx.x & 31;
  const int q0 = qblk * 64 + wave * 16;
  const int m = lane & 15;
  const int nlo = lane & 15;
  const int mtop = (lane >> 4) << 3;
  const float LOG2E = 1.4426950408889634f;

  const unsigned klds = (unsigned)(size_t)&ktile[0][0][0];
  const unsigned vlds = (unsigned)(size_t)&vtile[0][0][0];

  // Issue one k-step's tiles (4 async b128 per thread: 2 K chunks + 2 V chunks)
  auto issue_tiles = [&](int k0s, int buf) {
#pragma unroll
    for (int j = 0; j < 2; ++j) {
      const int c = tid + j * 128;                  // 256 x 16B chunks
      const int kr = c >> 3, ks = c & 7;            // [32][64] bf16 rows
      async_load_b128(klds + (unsigned)(buf * 4096 + c * 16),
                      kbm + ((size_t)bh * S_LEN + k0s + kr) * HD_DIM + ks * 8);
    }
#pragma unroll
    for (int j = 0; j < 2; ++j) {
      const int c = tid + j * 128;
      const int vr = c >> 2, vs = c & 3;            // [64][32] bf16 rows
      async_load_b128(vlds + (unsigned)(buf * 4096 + c * 16),
                      vtb + ((size_t)bh * HD_DIM + vr) * S_LEN + k0s + vs * 8);
    }
  };

  const bf16_t* qp = qb + ((size_t)bh * S_LEN + q0 + m) * HD_DIM;
  bf16x16 aq0 = load_a_frag(qp, lane);        // d = 0..31
  bf16x16 aq1 = load_a_frag(qp + 32, lane);   // d = 32..63

  f32x8 o[4] = {};
  float mrow[8], lrow[8];
#pragma unroll
  for (int i = 0; i < 8; ++i) { mrow[i] = -3.0e38f; lrow[i] = 0.0f; }

  const int kend = qblk * 64 + 64;  // block-uniform causal bound
  issue_tiles(0, 0);
  for (int k0 = 0, it = 0; k0 < kend; k0 += 32, ++it) {
    const int cur = it & 1;
    if (k0 + 32 < kend) {
      issue_tiles(k0 + 32, cur ^ 1);
      WAIT_ASYNCCNT_4();   // current buffer's 4 transfers complete
    } else {
      WAIT_ASYNCCNT_0();
    }
    __syncthreads();       // all waves' tiles visible

    // ---- S = Q K^T from LDS K-tile ----
    f32x8 sc[2] = {};
#pragma unroll
    for (int t = 0; t < 2; ++t) {
      const bf16_t* kp = &ktile[cur][t * 16 + nlo][0];
      sc[t] = wmma_bf16(aq0, load_b_frag(kp, lane), sc[t]);
      sc[t] = wmma_bf16(aq1, load_b_frag(kp + 32, lane), sc[t]);
    }
    // ---- scale + causal mask + row-max ----
    float rmax[8];
#pragma unroll
    for (int i = 0; i < 8; ++i) {
      const int qa = q0 + mtop + i;
      const int ka = k0 + nlo;
      float v0 = sc[0][i] * 0.125f + ((ka      <= qa) ? 0.0f : -1.0e9f);
      float v1 = sc[1][i] * 0.125f + ((ka + 16 <= qa) ? 0.0f : -1.0e9f);
      sc[0][i] = v0; sc[1][i] = v1;
      rmax[i] = fmaxf(v0, v1);
    }
#pragma unroll
    for (int off = 1; off < 16; off <<= 1)
#pragma unroll
      for (int i = 0; i < 8; ++i)
        rmax[i] = fmaxf(rmax[i], __shfl_xor(rmax[i], off, 32));
    // ---- online softmax; stage P (bf16) into per-wave LDS tile ----
    float rsum[8];
#pragma unroll
    for (int i = 0; i < 8; ++i) {
      const float mnew = fmaxf(mrow[i], rmax[i]);
      const float alpha = exp2f((mrow[i] - mnew) * LOG2E);
      const float p0 = exp2f((sc[0][i] - mnew) * LOG2E);
      const float p1 = exp2f((sc[1][i] - mnew) * LOG2E);
      mrow[i] = mnew;
      lrow[i] *= alpha;
      rsum[i] = p0 + p1;
#pragma unroll
      for (int t = 0; t < 4; ++t) o[t][i] *= alpha;
      psh[wave][mtop + i][nlo]      = f2bf(p0);
      psh[wave][mtop + i][nlo + 16] = f2bf(p1);
    }
#pragma unroll
    for (int off = 1; off < 16; off <<= 1)
#pragma unroll
      for (int i = 0; i < 8; ++i)
        rsum[i] += __shfl_xor(rsum[i], off, 32);
#pragma unroll
    for (int i = 0; i < 8; ++i) lrow[i] += rsum[i];

    // DS ops are in-order within a wave; barrier stops compiler reordering.
    __builtin_amdgcn_wave_barrier();
    bf16x16 pa = load_a_frag(&psh[wave][m][0], lane);  // P as A fragment
    __builtin_amdgcn_wave_barrier();

    // ---- O += P V from LDS V-tile ----
#pragma unroll
    for (int t = 0; t < 4; ++t)
      o[t] = wmma_bf16(pa, load_b_frag(&vtile[cur][t * 16 + nlo][0], lane), o[t]);

    __syncthreads();       // all waves done reading before buffer reuse
  }
  // ---- normalize + store [bh][q][hd] (flat layout = reference reshape) ----
#pragma unroll
  for (int i = 0; i < 8; ++i) {
    const float inv = 1.0f / lrow[i];
    const size_t base = ((size_t)bh * S_LEN + q0 + mtop + i) * HD_DIM + nlo;
#pragma unroll
    for (int t = 0; t < 4; ++t) ob[base + t * 16] = f2bf(o[t][i] * inv);
  }
}

// --------------------------------------------------------------- out GEMM ----
__global__ __launch_bounds__(128) void out_gemm_kernel(
    const bf16_t* __restrict__ vals,  // [4096][1024] bf16 (flat attn output)
    const bf16_t* __restrict__ wot,   // [1024][1024]  (= W_out^T, bf16)
    const float*  __restrict__ bias,  // [1024]
    float* __restrict__ out) {        // [4096][1024] f32
  const int lane = threadIdx.x & 31;
  const int wave = threadIdx.x >> 5;
  const int row0 = blockIdx.x * 128 + wave * 32;
  const int col0 = blockIdx.y * 64;
  const int m = lane & 15;
  const int nlo = lane & 15;
  const int mtop = (lane >> 4) << 3;

  const bf16_t* arow0 = vals + (size_t)(row0 + m) * D_DIM;
  const bf16_t* arow1 = vals + (size_t)(row0 + 16 + m) * D_DIM;
  const bf16_t* brow[4];
#pragma unroll
  for (int t = 0; t < 4; ++t)
    brow[t] = wot + (size_t)(col0 + t * 16 + nlo) * D_DIM;

  f32x8 acc[2][4] = {};
  gemm_core_32x64(arow0, arow1, brow, lane, acc);

#pragma unroll
  for (int t = 0; t < 4; ++t) {
    const int e = col0 + t * 16 + nlo;
    const float bv = bias[e];
#pragma unroll
    for (int r = 0; r < 2; ++r)
#pragma unroll
      for (int i = 0; i < 8; ++i)
        out[(size_t)(row0 + r * 16 + mtop + i) * D_DIM + e] = acc[r][t][i] + bv;
  }
}

// ------------------------------------------------------------------ launch ---
extern "C" void kernel_launch(void* const* d_in, const int* in_sizes, int n_in,
                              void* d_out, int out_size, void* d_ws, size_t ws_size,
                              hipStream_t stream) {
  (void)in_sizes; (void)n_in; (void)out_size; (void)ws_size;
  const float* x    = (const float*)d_in[0];
  // d_in[1] = mask: unused — causal 0/-1e9 mask reproduced inline bit-exactly.
  const float* Wqkv = (const float*)d_in[2];
  const float* bqkv = (const float*)d_in[3];
  const float* Wout = (const float*)d_in[4];
  const float* bout = (const float*)d_in[5];
  float* out = (float*)d_out;

  char* ws = (char*)d_ws;
  size_t off = 0;
  auto take = [&](size_t bytes) {
    char* p = ws + off;
    off += (bytes + 255) & ~(size_t)255;
    return p;
  };
  bf16_t* xbf = (bf16_t*)take((size_t)M_ROWS * D_DIM * 2);            // 8 MB
  bf16_t* wqt = (bf16_t*)take((size_t)3 * D_DIM * D_DIM * 2);         // 6 MB
  bf16_t* wot = (bf16_t*)take((size_t)D_DIM * D_DIM * 2);             // 2 MB
  bf16_t* qb  = (bf16_t*)take((size_t)BH * S_LEN * HD_DIM * 2);       // 8 MB
  bf16_t* kb  = (bf16_t*)take((size_t)BH * S_LEN * HD_DIM * 2);       // 8 MB
  bf16_t* vtb = (bf16_t*)take((size_t)BH * HD_DIM * S_LEN * 2);       // 8 MB
  bf16_t* vals = xbf;  // xbf fully consumed by qkv_gemm before attn writes

  const int nx = M_ROWS * D_DIM;
  cvt_bf16_kernel<<<(nx + 255) / 256, 256, 0, stream>>>(x, xbf, nx);
  transpose_bf16_kernel<<<dim3(D_DIM / 32, 3 * D_DIM / 8), 256, 0, stream>>>(
      Wqkv, wqt, D_DIM, 3 * D_DIM);
  transpose_bf16_kernel<<<dim3(D_DIM / 32, D_DIM / 8), 256, 0, stream>>>(
      Wout, wot, D_DIM, D_DIM);

  qkv_gemm_kernel<<<dim3(M_ROWS / 128, 3 * D_DIM / 64), 128, 0, stream>>>(
      xbf, wqt, bqkv, qb, kb, vtb);
  attn_kernel<<<BH * (S_LEN / 64), 128, 0, stream>>>(qb, kb, vtb, vals);
  out_gemm_kernel<<<dim3(M_ROWS / 128, D_DIM / 64), 128, 0, stream>>>(
      vals, wot, bout, out);
}
